// SSCDNonLModel_gcn_a_62594853372072
// MI455X (gfx1250) — matrix operand order repaired and checked by reading, not compiled, not run on hardware
//
#include <hip/hip_runtime.h>

#define NPIX (512 * 512)

typedef __attribute__((ext_vector_type(16))) __bf16 v16bf;
typedef __attribute__((ext_vector_type(8)))  float v8f;

union BfPack { v16bf v; unsigned short u[16]; };

__device__ __forceinline__ unsigned short f2bf(float f) {
  union { float f; unsigned int u; } c; c.f = f;
  unsigned int r = c.u + 0x7FFFu + ((c.u >> 16) & 1u);  // round-to-nearest-even
  return (unsigned short)(r >> 16);
}

__device__ __forceinline__ float sigmoidf(float v) {
  return 1.0f / (1.0f + __expf(-v));
}

// ---------------------------------------------------------------------------
// BN folding helpers: BN(x) = s*x + t per channel; conv(BN(x)) folds into
// W' = W*s (bf16) and b' = b + W@t.
// ---------------------------------------------------------------------------
__global__ void make_affine(const float* __restrict__ g, const float* __restrict__ b,
                            const float* __restrict__ rm, const float* __restrict__ rv,
                            float* __restrict__ s, float* __restrict__ t, int C) {
  int i = blockIdx.x * blockDim.x + threadIdx.x;
  if (i < C) {
    float sc = g[i] * rsqrtf(rv[i] + 1e-5f);
    s[i] = sc;
    t[i] = b[i] - rm[i] * sc;
  }
}

// con4 sees BN_bcat(concat(BN_b4(R), G2)): compose the two affines on ch 0..24
__global__ void build_con4_affine(const float* __restrict__ sb, const float* __restrict__ tb,
                                  const float* __restrict__ s4, const float* __restrict__ t4,
                                  float* __restrict__ s, float* __restrict__ t) {
  int i = threadIdx.x;
  if (i < 50) {
    if (i < 25) { s[i] = sb[i] * s4[i]; t[i] = sb[i] * t4[i] + tb[i]; }
    else        { s[i] = sb[i];          t[i] = tb[i]; }
  }
}

__global__ void fold_w(const float* __restrict__ Wsrc, const float* __restrict__ s,
                       unsigned short* __restrict__ Wf, int M, int K, int Kpad, int Mpad) {
  int idx = blockIdx.x * 256 + threadIdx.x;
  if (idx >= Mpad * Kpad) return;
  int m = idx / Kpad, k = idx % Kpad;
  float v = 0.0f;
  if (m < M && k < K) v = Wsrc[m * K + k] * (s ? s[k] : 1.0f);
  Wf[idx] = f2bf(v);
}

__global__ void fold_b(const float* __restrict__ cb, const float* __restrict__ Wsrc,
                       const float* __restrict__ t, float* __restrict__ bout, int M, int K) {
  int m = blockIdx.x * blockDim.x + threadIdx.x;
  if (m >= M) return;
  float acc = cb[m];
  if (t) for (int k = 0; k < K; ++k) acc += Wsrc[m * K + k] * t[k];
  bout[m] = acc;
}

// ---------------------------------------------------------------------------
// Fused 1x1-conv GEMM: out[oc,p] = act( sum_k W'[oc,k] * B[k,p] + b'[oc] )
// B rows 0..C1-1 come from src1, C1..C1+C2-1 from src2 (virtual concat),
// rows >= C1+C2 are zero padding.  bf16 WMMA 16x16x32, fp32 accumulate.
// One wave owns one 16-pixel tile and ALL MT m-tiles (B fetched once).
// K-loop is segmented: all-src1 / all-src2 blocks are wave-uniform branchless
// fast paths; only the concat-seam and zero-tail blocks pay per-element selects.
// ---------------------------------------------------------------------------
template <int MT>
__global__ __launch_bounds__(256) void conv1x1_wmma(
    const unsigned short* __restrict__ Wf, const float* __restrict__ bias,
    const float* __restrict__ src1, int C1,
    const float* __restrict__ src2, int C2,
    float* __restrict__ out, int M, int Kpad, int applySig) {
  const int lane  = threadIdx.x & 31;
  const int wave  = threadIdx.x >> 5;
  const int laneL = lane & 15;
  const bool hi   = lane >= 16;
  const int ptile = blockIdx.x * 8 + wave;
  const int p     = ptile * 16 + laneL;
  const int Ctot  = C1 + C2;

  v8f acc[MT];
#pragma unroll
  for (int mt = 0; mt < MT; ++mt) {
    v8f z = {0.f, 0.f, 0.f, 0.f, 0.f, 0.f, 0.f, 0.f};
    acc[mt] = z;
  }

  __builtin_prefetch(Wf + (size_t)laneL * Kpad, 0, 3);

  for (int kb = 0; kb < Kpad; kb += 32) {
    BfPack B;
    const int kofs = kb + (hi ? 16 : 0);
    if (kb + 32 <= C1) {
      // fast path: whole block from src1 (wave-uniform branch, clean clause)
      const float* rp = src1 + (size_t)kofs * NPIX + p;
#pragma unroll
      for (int h = 0; h < 16; ++h) B.u[h] = f2bf(rp[(size_t)h * NPIX]);
    } else if (kb >= C1 && kb + 32 <= Ctot) {
      // fast path: whole block from src2
      const float* rp = src2 + (size_t)(kofs - C1) * NPIX + p;
#pragma unroll
      for (int h = 0; h < 16; ++h) B.u[h] = f2bf(rp[(size_t)h * NPIX]);
    } else {
      // concat seam / zero tail (at most two blocks per GEMM)
#pragma unroll
      for (int h = 0; h < 16; ++h) {
        const int k = kofs + h;
        float v = 0.0f;
        if (k < C1)        v = src1[(size_t)k * NPIX + p];
        else if (k < Ctot) v = src2[(size_t)(k - C1) * NPIX + p];
        B.u[h] = f2bf(v);
      }
    }
#pragma unroll
    for (int mt = 0; mt < MT; ++mt) {
      // 16-bit A 16x32: lane L holds row M=L; halves 0..15 -> K+0..15 (hi:+16)
      BfPack A;
      const unsigned short* wr =
          Wf + (size_t)(mt * 16 + laneL) * Kpad + kb + (hi ? 16 : 0);
#pragma unroll
      for (int h = 0; h < 16; ++h) { A.u[h] = wr[h]; }
      acc[mt] = __builtin_amdgcn_wmma_f32_16x16x32_bf16(
          false, A.v, false, B.v, (short)0, acc[mt], false, false);
    }
  }

#pragma unroll
  for (int mt = 0; mt < MT; ++mt) {
#pragma unroll
    for (int r = 0; r < 8; ++r) {
      const int oc = mt * 16 + r + (hi ? 8 : 0);
      if (oc < M) {
        float v = acc[mt][r] + bias[oc];
        if (applySig) v = sigmoidf(v);
        out[(size_t)oc * NPIX + p] = v;
      }
    }
  }
}

// ---------------------------------------------------------------------------
// Grouped 5x5 conv (pad 2) + sigmoid.  cpg=1 -> depthwise.  Channel-major.
// ---------------------------------------------------------------------------
__global__ void grouped_conv5x5_sig(const float* __restrict__ in,
                                    const float* __restrict__ w,
                                    const float* __restrict__ b,
                                    float* __restrict__ out, int G, int cpg) {
  const size_t idx = (size_t)blockIdx.x * 256 + threadIdx.x;
  const int gc  = (int)(idx >> 18);
  const int pix = (int)(idx & (NPIX - 1));
  const int y = pix >> 9, x = pix & 511;
  float acc = b[gc];
  for (int ci = 0; ci < cpg; ++ci) {
    const float* ip = in + ((size_t)(gc * cpg + ci) << 18);
    const float* wp = w + (gc * cpg + ci) * 25;
#pragma unroll
    for (int dy = -2; dy <= 2; ++dy) {
      const int yy = y + dy;
      if (yy < 0 || yy >= 512) continue;
#pragma unroll
      for (int dx = -2; dx <= 2; ++dx) {
        const int xx = x + dx;
        if (xx < 0 || xx >= 512) continue;
        acc += ip[yy * 512 + xx] * wp[(dy + 2) * 5 + (dx + 2)];
      }
    }
  }
  out[idx] = sigmoidf(acc);
}

// ---------------------------------------------------------------------------
// att / trans 1x1 convs on xx = BN_b4(R), with b4 folded inline.
// ---------------------------------------------------------------------------
__global__ void att_trans_kernel(const float* __restrict__ R,
                                 const float* __restrict__ s4, const float* __restrict__ t4,
                                 const float* __restrict__ attw, const float* __restrict__ attb,
                                 const float* __restrict__ trw,  const float* __restrict__ trb,
                                 float* __restrict__ aOut, float* __restrict__ tOut) {
  const int pix = blockIdx.x * 256 + threadIdx.x;
  float a = attb[0], t = trb[0];
#pragma unroll
  for (int c = 0; c < 25; ++c) {
    const float xb = R[(size_t)c * NPIX + pix] * s4[c] + t4[c];
    a += attw[c] * xb;
    t += trw[c] * xb;
  }
  aOut[pix] = a;
  tOut[pix] = t;
}

// --------------------------- deterministic reductions ----------------------
__global__ void partial_max_kernel(const float* __restrict__ a, float* __restrict__ part) {
  __shared__ float sm[256];
  const int pix = blockIdx.x * 256 + threadIdx.x;
  sm[threadIdx.x] = a[pix];
  __syncthreads();
  for (int s = 128; s > 0; s >>= 1) {
    if (threadIdx.x < s) sm[threadIdx.x] = fmaxf(sm[threadIdx.x], sm[threadIdx.x + s]);
    __syncthreads();
  }
  if (threadIdx.x == 0) part[blockIdx.x] = sm[0];
}

__global__ void final_max_kernel(const float* __restrict__ part, int n, float* __restrict__ dst) {
  __shared__ float sm[256];
  float m = -3.402823466e+38f;
  for (int i = threadIdx.x; i < n; i += 256) m = fmaxf(m, part[i]);
  sm[threadIdx.x] = m;
  __syncthreads();
  for (int s = 128; s > 0; s >>= 1) {
    if (threadIdx.x < s) sm[threadIdx.x] = fmaxf(sm[threadIdx.x], sm[threadIdx.x + s]);
    __syncthreads();
  }
  if (threadIdx.x == 0) dst[0] = sm[0];
}

__global__ void final_sum_kernel(const float* __restrict__ part, int n, float* __restrict__ dst) {
  __shared__ float sm[256];
  float a = 0.0f;
  for (int i = threadIdx.x; i < n; i += 256) a += part[i];
  sm[threadIdx.x] = a;
  __syncthreads();
  for (int s = 128; s > 0; s >>= 1) {
    if (threadIdx.x < s) sm[threadIdx.x] += sm[threadIdx.x + s];
    __syncthreads();
  }
  if (threadIdx.x == 0) dst[0] = sm[0];
}

__global__ void partial_sumexp_kernel(const float* __restrict__ a, const float* __restrict__ SC,
                                      float* __restrict__ part) {
  __shared__ float sm[256];
  const int pix = blockIdx.x * 256 + threadIdx.x;
  sm[threadIdx.x] = __expf(a[pix] - SC[0]);
  __syncthreads();
  for (int s = 128; s > 0; s >>= 1) {
    if (threadIdx.x < s) sm[threadIdx.x] += sm[threadIdx.x + s];
    __syncthreads();
  }
  if (threadIdx.x == 0) part[blockIdx.x] = sm[0];
}

// mask_i = (softmax_i != 0);  partials of S = sum(mask), Tm = sum(mask * t)
__global__ void partial_mask_stats(const float* __restrict__ a, const float* __restrict__ t,
                                   const float* __restrict__ SC,
                                   float* __restrict__ pS, float* __restrict__ pT) {
  __shared__ float sm[256];
  const int pix = blockIdx.x * 256 + threadIdx.x;
  const float mask = (__expf(a[pix] - SC[0]) / SC[1] != 0.0f) ? 1.0f : 0.0f;
  sm[threadIdx.x] = mask;
  __syncthreads();
  for (int s = 128; s > 0; s >>= 1) {
    if (threadIdx.x < s) sm[threadIdx.x] += sm[threadIdx.x + s];
    __syncthreads();
  }
  if (threadIdx.x == 0) pS[blockIdx.x] = sm[0];
  __syncthreads();
  sm[threadIdx.x] = mask * t[pix];
  __syncthreads();
  for (int s = 128; s > 0; s >>= 1) {
    if (threadIdx.x < s) sm[threadIdx.x] += sm[threadIdx.x + s];
    __syncthreads();
  }
  if (threadIdx.x == 0) pT[blockIdx.x] = sm[0];
}

// SC: [0]=amax [1]=sumexp [2]=S [3]=Tm [4]=dinv0 [5]=mask0 [6..17]=g1 node0
__global__ void gcn_node0(const float* __restrict__ tbuf, const float* __restrict__ abuf,
                          const float* __restrict__ g1w, const float* __restrict__ g1b,
                          float* __restrict__ SC) {
  const float amax = SC[0], sumexp = SC[1], S = SC[2], Tm = SC[3];
  const float mask0 = (__expf(abuf[0] - amax) / sumexp != 0.0f) ? 1.0f : 0.0f;
  const float dinv0 = rsqrtf(1.0f + S);
  const float t0 = tbuf[0];
  const float corr = (Tm - mask0 * t0) + mask0 * dinv0 * t0;  // sum_i mask*dinv*t
#pragma unroll
  for (int f = 0; f < 12; ++f) {
    const float v = dinv0 * dinv0 * t0 * g1w[f] + dinv0 * g1w[f] * corr + g1b[f];
    SC[6 + f] = fmaxf(v, 0.0f);
  }
  SC[4] = dinv0;
  SC[5] = mask0;
}

// Per node i: g1_i = relu(t_i*w + b); h'_i = g1_i @ W2.
// For i>0 write G2flat[i*25+c] = h' + b2 (dinv=1).  Block-reduce mask*h' -> pRc.
// G2flat [N*25] row-major == the raw-reshaped [25,H,W] channel-major tensor.
__global__ void gcn_map(const float* __restrict__ tbuf, const float* __restrict__ abuf,
                        const float* __restrict__ SC,
                        const float* __restrict__ g1w, const float* __restrict__ g1b,
                        const float* __restrict__ g2w, const float* __restrict__ g2b,
                        float* __restrict__ G2, float* __restrict__ pRc) {
  __shared__ float sm[256];
  const int pix = blockIdx.x * 256 + threadIdx.x;
  const float ti = tbuf[pix];
  float g1v[12];
#pragma unroll
  for (int f = 0; f < 12; ++f) g1v[f] = fmaxf(ti * g1w[f] + g1b[f], 0.0f);
  const float mask = (__expf(abuf[pix] - SC[0]) / SC[1] != 0.0f) ? 1.0f : 0.0f;
  const float contrib = (pix == 0) ? 0.0f : mask;  // node 0 folded in later
  for (int c = 0; c < 25; ++c) {
    float v = 0.0f;
#pragma unroll
    for (int f = 0; f < 12; ++f) v += g1v[f] * g2w[f * 25 + c];
    if (pix != 0) G2[(size_t)pix * 25 + c] = v + g2b[c];
    sm[threadIdx.x] = contrib * v;
    __syncthreads();
    for (int s = 128; s > 0; s >>= 1) {
      if (threadIdx.x < s) sm[threadIdx.x] += sm[threadIdx.x + s];
      __syncthreads();
    }
    if (threadIdx.x == 0) pRc[blockIdx.x * 25 + c] = sm[0];
    __syncthreads();
  }
}

// Reduce pRc, then write node-0 row (with aggregated message) into G2[0..24].
__global__ void finish_gcn(const float* __restrict__ pRc, int nblk,
                           const float* __restrict__ SC,
                           const float* __restrict__ g2w, const float* __restrict__ g2b,
                           float* __restrict__ G2) {
  const int c = threadIdx.x;
  if (c >= 25) return;
  float rc = 0.0f;
  for (int b = 0; b < nblk; ++b) rc += pRc[b * 25 + c];
  const float dinv0 = SC[4], mask0 = SC[5];
  float h0 = 0.0f;
#pragma unroll
  for (int f = 0; f < 12; ++f) h0 += SC[6 + f] * g2w[f * 25 + c];
  const float msg2 = dinv0 * (rc + mask0 * dinv0 * h0);
  G2[c] = dinv0 * dinv0 * h0 + msg2 + g2b[c];
}

// ---------------------------------------------------------------------------
extern "C" void kernel_launch(void* const* d_in, const int* in_sizes, int n_in,
                              void* d_out, int out_size, void* d_ws, size_t ws_size,
                              hipStream_t stream) {
  (void)n_in; (void)out_size; (void)ws_size;
  const float* const* F = (const float* const*)d_in;

  struct P {
    const float *x, *b1g, *b1b, *b1rm, *b1rv, *con1w, *con1b, *cond1w, *cond1b,
        *b2g, *b2b, *b2rm, *b2rv, *con2w, *con2b, *cond2w, *cond2b,
        *b4g, *b4b, *b4rm, *b4rv, *attw, *attb, *trw, *trb,
        *g1w, *g1b, *g2w, *g2b, *bcg, *bcb, *bcrm, *bcrv,
        *con4w, *con4b, *cond4w, *cond4b, *b5g, *b5b, *b5rm, *b5rv,
        *con5w, *con5b, *cond5w, *cond5b, *con6w, *con6b;
  } p;

  if (in_sizes[0] == 103 * NPIX) {
    // insertion-order flatten: x first, params in construction order (g,b,rm,rv)
    p.x = F[0];
    p.b1g = F[1];  p.b1b = F[2];  p.b1rm = F[3];  p.b1rv = F[4];
    p.con1w = F[5];  p.con1b = F[6];  p.cond1w = F[7];  p.cond1b = F[8];
    p.b2g = F[9];  p.b2b = F[10]; p.b2rm = F[11]; p.b2rv = F[12];
    p.con2w = F[13]; p.con2b = F[14]; p.cond2w = F[15]; p.cond2b = F[16];
    p.b4g = F[17]; p.b4b = F[18]; p.b4rm = F[19]; p.b4rv = F[20];
    p.attw = F[21]; p.attb = F[22]; p.trw = F[23]; p.trb = F[24];
    p.g1w = F[25]; p.g1b = F[26]; p.g2w = F[27]; p.g2b = F[28];
    p.bcg = F[29]; p.bcb = F[30]; p.bcrm = F[31]; p.bcrv = F[32];
    p.con4w = F[33]; p.con4b = F[34]; p.cond4w = F[35]; p.cond4b = F[36];
    p.b5g = F[37]; p.b5b = F[38]; p.b5rm = F[39]; p.b5rv = F[40];
    p.con5w = F[41]; p.con5b = F[42]; p.cond5w = F[43]; p.cond5b = F[44];
    p.con6w = F[45]; p.con6b = F[46];
  } else {
    // JAX tree-flatten (sorted keys): params first (keys sorted, bn = b,g,rm,rv), then x
    p.attb = F[0]; p.attw = F[1];
    p.b1b = F[2];  p.b1g = F[3];  p.b1rm = F[4];  p.b1rv = F[5];
    p.b2b = F[6];  p.b2g = F[7];  p.b2rm = F[8];  p.b2rv = F[9];
    p.b4b = F[10]; p.b4g = F[11]; p.b4rm = F[12]; p.b4rv = F[13];
    p.b5b = F[14]; p.b5g = F[15]; p.b5rm = F[16]; p.b5rv = F[17];
    p.bcb = F[18]; p.bcg = F[19]; p.bcrm = F[20]; p.bcrv = F[21];
    p.con1b = F[22]; p.con1w = F[23]; p.con2b = F[24]; p.con2w = F[25];
    p.con4b = F[26]; p.con4w = F[27]; p.con5b = F[28]; p.con5w = F[29];
    p.con6b = F[30]; p.con6w = F[31];
    p.cond1b = F[32]; p.cond1w = F[33]; p.cond2b = F[34]; p.cond2w = F[35];
    p.cond4b = F[36]; p.cond4w = F[37]; p.cond5b = F[38]; p.cond5w = F[39];
    p.g1b = F[40]; p.g1w = F[41]; p.g2b = F[42]; p.g2w = F[43];
    p.trb = F[44]; p.trw = F[45]; p.x = F[46];
  }

  // ---- workspace carve-up ----
  char* base = (char*)d_ws;
  size_t off = 0;
  auto alloc = [&](size_t bytes) -> void* {
    void* q = base + off;
    off += (bytes + 255) & ~(size_t)255;
    return q;
  };
  float* P0 = (float*)alloc((size_t)100 * NPIX * 4);  // ping
  float* Q0 = (float*)alloc((size_t)100 * NPIX * 4);  // pong
  float* R  = (float*)alloc((size_t)25 * NPIX * 4);   // out2 (25ch), live to end
  float* G2 = (float*)alloc((size_t)25 * NPIX * 4);   // gcn output, flat==channel-major
  float* tbuf = (float*)alloc((size_t)NPIX * 4);
  float* abuf = (float*)alloc((size_t)NPIX * 4);
  float* part1 = (float*)alloc(1024 * 4);
  float* part2 = (float*)alloc(1024 * 4);
  float* pRc   = (float*)alloc(1024 * 25 * 4);
  float* SC    = (float*)alloc(64 * 4);
  float* s1 = (float*)alloc(103 * 4); float* t1 = (float*)alloc(103 * 4);
  float* s2 = (float*)alloc(203 * 4); float* t2 = (float*)alloc(203 * 4);
  float* s4 = (float*)alloc(25 * 4);  float* t4 = (float*)alloc(25 * 4);
  float* sb = (float*)alloc(50 * 4);  float* tb = (float*)alloc(50 * 4);
  float* sc4 = (float*)alloc(50 * 4); float* tc4 = (float*)alloc(50 * 4);
  float* s5 = (float*)alloc(125 * 4); float* t5 = (float*)alloc(125 * 4);
  unsigned short* Wf1 = (unsigned short*)alloc(112 * 128 * 2);
  unsigned short* Wf2 = (unsigned short*)alloc(112 * 224 * 2);
  unsigned short* Wf4 = (unsigned short*)alloc(112 * 64 * 2);
  unsigned short* Wf5 = (unsigned short*)alloc(112 * 128 * 2);
  unsigned short* Wf6 = (unsigned short*)alloc(16 * 128 * 2);
  float* Bf1 = (float*)alloc(112 * 4); float* Bf2 = (float*)alloc(112 * 4);
  float* Bf4 = (float*)alloc(112 * 4); float* Bf5 = (float*)alloc(112 * 4);
  float* Bf6 = (float*)alloc(16 * 4);

  // ---- BN affines + weight folding (tiny) ----
  make_affine<<<1, 128, 0, stream>>>(p.b1g, p.b1b, p.b1rm, p.b1rv, s1, t1, 103);
  make_affine<<<1, 256, 0, stream>>>(p.b2g, p.b2b, p.b2rm, p.b2rv, s2, t2, 203);
  make_affine<<<1, 32, 0, stream>>>(p.b4g, p.b4b, p.b4rm, p.b4rv, s4, t4, 25);
  make_affine<<<1, 64, 0, stream>>>(p.bcg, p.bcb, p.bcrm, p.bcrv, sb, tb, 50);
  make_affine<<<1, 128, 0, stream>>>(p.b5g, p.b5b, p.b5rm, p.b5rv, s5, t5, 125);
  build_con4_affine<<<1, 64, 0, stream>>>(sb, tb, s4, t4, sc4, tc4);

  fold_w<<<(112 * 128 + 255) / 256, 256, 0, stream>>>(p.con1w, s1, Wf1, 100, 103, 128, 112);
  fold_b<<<1, 128, 0, stream>>>(p.con1b, p.con1w, t1, Bf1, 100, 103);
  fold_w<<<(112 * 224 + 255) / 256, 256, 0, stream>>>(p.con2w, s2, Wf2, 100, 203, 224, 112);
  fold_b<<<1, 128, 0, stream>>>(p.con2b, p.con2w, t2, Bf2, 100, 203);
  fold_w<<<(112 * 64 + 255) / 256, 256, 0, stream>>>(p.con4w, sc4, Wf4, 100, 50, 64, 112);
  fold_b<<<1, 128, 0, stream>>>(p.con4b, p.con4w, tc4, Bf4, 100, 50);
  fold_w<<<(112 * 128 + 255) / 256, 256, 0, stream>>>(p.con5w, s5, Wf5, 100, 125, 128, 112);
  fold_b<<<1, 128, 0, stream>>>(p.con5b, p.con5w, t5, Bf5, 100, 125);
  fold_w<<<(16 * 128 + 255) / 256, 256, 0, stream>>>(p.con6w, nullptr, Wf6, 10, 125, 128, 16);
  fold_b<<<1, 32, 0, stream>>>(p.con6b, p.con6w, nullptr, Bf6, 10, 125);

  const int gemmBlocks = NPIX / 128;  // 8 waves/block, 16 px/wave

  // ---- main pipeline ----
  conv1x1_wmma<7><<<gemmBlocks, 256, 0, stream>>>(Wf1, Bf1, p.x, 103, nullptr, 0, P0, 100, 128, 1);
  grouped_conv5x5_sig<<<(100 * NPIX) / 256, 256, 0, stream>>>(P0, p.cond1w, p.cond1b, Q0, 100, 1);
  conv1x1_wmma<7><<<gemmBlocks, 256, 0, stream>>>(Wf2, Bf2, Q0, 100, p.x, 103, P0, 100, 224, 1);
  grouped_conv5x5_sig<<<(25 * NPIX) / 256, 256, 0, stream>>>(P0, p.cond2w, p.cond2b, R, 25, 4);

  // ---- attention + GCN ----
  att_trans_kernel<<<NPIX / 256, 256, 0, stream>>>(R, s4, t4, p.attw, p.attb, p.trw, p.trb, abuf, tbuf);
  partial_max_kernel<<<NPIX / 256, 256, 0, stream>>>(abuf, part1);
  final_max_kernel<<<1, 256, 0, stream>>>(part1, NPIX / 256, SC + 0);
  partial_sumexp_kernel<<<NPIX / 256, 256, 0, stream>>>(abuf, SC, part1);
  final_sum_kernel<<<1, 256, 0, stream>>>(part1, NPIX / 256, SC + 1);
  partial_mask_stats<<<NPIX / 256, 256, 0, stream>>>(abuf, tbuf, SC, part1, part2);
  final_sum_kernel<<<1, 256, 0, stream>>>(part1, NPIX / 256, SC + 2);
  final_sum_kernel<<<1, 256, 0, stream>>>(part2, NPIX / 256, SC + 3);
  gcn_node0<<<1, 1, 0, stream>>>(tbuf, abuf, p.g1w, p.g1b, SC);
  gcn_map<<<NPIX / 256, 256, 0, stream>>>(tbuf, abuf, SC, p.g1w, p.g1b, p.g2w, p.g2b, G2, pRc);
  finish_gcn<<<1, 32, 0, stream>>>(pRc, NPIX / 256, SC, p.g2w, p.g2b, G2);

  // ---- tail ----
  conv1x1_wmma<7><<<gemmBlocks, 256, 0, stream>>>(Wf4, Bf4, R, 25, G2, 25, P0, 100, 64, 1);
  grouped_conv5x5_sig<<<(100 * NPIX) / 256, 256, 0, stream>>>(P0, p.cond4w, p.cond4b, Q0, 100, 1);
  conv1x1_wmma<7><<<gemmBlocks, 256, 0, stream>>>(Wf5, Bf5, Q0, 100, R, 25, P0, 100, 128, 1);
  grouped_conv5x5_sig<<<(100 * NPIX) / 256, 256, 0, stream>>>(P0, p.cond5w, p.cond5b, Q0, 100, 1);
  conv1x1_wmma<1><<<gemmBlocks, 256, 0, stream>>>(Wf6, Bf6, Q0, 100, R, 25, (float*)d_out, 10, 128, 0);
}
